// Model_54107997995563
// MI455X (gfx1250) — matrix operand order repaired and checked by reading, not compiled, not run on hardware
//
#include <hip/hip_runtime.h>
#include <stdint.h>

// ---------------- CDNA5 WMMA types ----------------
typedef __attribute__((ext_vector_type(16))) __bf16 v16bf;
typedef __attribute__((ext_vector_type(8)))  float  v8f;

union FragAB { v16bf v; uint4 q[2]; };

// ---------------- problem constants (from reference) ----------------
#define NWAIT   20000
#define NPICK   100000
#define NTRANS  100000
#define NMOVE   100000
#define OUTC    256
#define HIDC    512

// ---------------- helpers ----------------
__device__ __forceinline__ unsigned int packbf2(float a, float b) {
  unsigned int ua = __float_as_uint(a);
  unsigned int ub = __float_as_uint(b);
  ua = (ua + 0x7FFFu + ((ua >> 16) & 1u)) >> 16;
  ub = (ub + 0x7FFFu + ((ub >> 16) & 1u)) >> 16;
  return (ua & 0xFFFFu) | (ub << 16);
}
__device__ __forceinline__ unsigned short f2bf(float a) {
  unsigned int ua = __float_as_uint(a);
  return (unsigned short)((ua + 0x7FFFu + ((ua >> 16) & 1u)) >> 16);
}

// ---------------- wait broadcast: out rows [0, NWAIT) ----------------
__global__ void wait_bcast_kernel(const float* __restrict__ w, float* __restrict__ out) {
  out[(size_t)blockIdx.x * OUTC + threadIdx.x] = w[threadIdx.x];
}

// ---------------- weight transpose + f32->bf16: Wt[n*K + k] = bf16(W[k*N + n]) ----------------
__global__ void wtrans_kernel(const float* __restrict__ W, unsigned short* __restrict__ Wt,
                              int K, int N) {
  int id = blockIdx.x * 256 + threadIdx.x;
  if (id >= K * N) return;
  int k = id / N, n = id - k * N;
  Wt[(size_t)n * K + k] = f2bf(W[id]);
}

// ---------------- pick gather -> bf16 X [NPICK x 1024] ----------------
__global__ void pick_gather_kernel(const float* __restrict__ agv, const float* __restrict__ op,
                                   const float* __restrict__ mach, const int* __restrict__ idx,
                                   unsigned short* __restrict__ X) {
  int row = blockIdx.x;
  int4 id = *(const int4*)(idx + (size_t)row * 4);
  int t = threadIdx.x;
  int c = t * 4;            // 4 floats per thread, 256 threads = 1024 cols
  int seg = c >> 8;
  int off = c & 255;
  const float* src;
  if (seg == 0)      src = agv  + (size_t)id.x * 256;
  else if (seg == 1) src = op   + (size_t)id.y * 256;
  else if (seg == 2) src = op   + (size_t)id.z * 256;
  else               src = mach + (size_t)id.w * 256;
  float4 v = *(const float4*)(src + off);
  uint2 o;
  o.x = packbf2(v.x, v.y);
  o.y = packbf2(v.z, v.w);
  *(uint2*)(X + (size_t)row * 1024 + c) = o;
}

// ---------------- transport/move gather -> f32 X [N x 512] ----------------
__global__ void pair_gather_kernel(const float* __restrict__ agv, const float* __restrict__ mach,
                                   const int* __restrict__ idx, float* __restrict__ X, int Nrows) {
  int t = threadIdx.x;
  int row = blockIdx.x * 2 + (t >> 7);
  if (row >= Nrows) return;
  int lt = t & 127;
  int c = lt * 4;            // 4 floats per thread, 128 threads per row
  int2 id = *(const int2*)(idx + (size_t)row * 2);
  const float* src = (c < 256) ? (agv + (size_t)id.x * 256 + c)
                               : (mach + (size_t)id.y * 256 + (c - 256));
  *(float4*)(X + (size_t)row * 512 + c) = *(const float4*)src;
}

// ---------------- tiled WMMA GEMM: C = act(A @ Bt^T + bias) ----------------
// A: M x K (f32 or bf16 row-major).  Bt: N x K bf16 (pre-transposed weights).
// Block tile 128x128, K-step 32, 8 waves (wave32), wave tile 32x64.
template<int A_BF16, int ACT_TANH>
__global__ __launch_bounds__(256)
void wmma_gemm_kernel(const void* __restrict__ Aptr,
                      const unsigned short* __restrict__ Bt,
                      const float* __restrict__ bias,
                      float* __restrict__ C,
                      int M, int N, int K, int ldc) {
  __shared__ unsigned short As[128 * 40];  // +8 bf16 pad per row: conflict-free frag reads
  __shared__ unsigned short Bs[128 * 40];

  const int tid   = threadIdx.x;
  const int lane  = tid & 31;
  const int wave  = tid >> 5;
  const int waveM = wave & 3;   // 4 waves along M (32 rows each)
  const int waveN = wave >> 2;  // 2 waves along N (64 cols each)
  const int bm = blockIdx.y * 128;
  const int bn = blockIdx.x * 128;
  const int lm = lane & 15;
  const int lh = lane >> 4;

  v8f acc[2][4];
  const v8f vzero = {0.f, 0.f, 0.f, 0.f, 0.f, 0.f, 0.f, 0.f};
#pragma unroll
  for (int i = 0; i < 2; ++i)
#pragma unroll
    for (int j = 0; j < 4; ++j) acc[i][j] = vzero;

  for (int k0 = 0; k0 < K; k0 += 32) {
    // ---- stage A tile 128x32 (convert f32->bf16 if needed) ----
    {
      const int r  = tid >> 1;
      const int kh = (tid & 1) << 4;  // 0 or 16 elements
      const int grow = bm + r;
      uint4 d0 = {0u, 0u, 0u, 0u}, d1 = {0u, 0u, 0u, 0u};
      if (A_BF16) {
        if (grow < M) {
          const uint4* src = (const uint4*)((const unsigned short*)Aptr + (size_t)grow * K + k0 + kh);
          d0 = src[0]; d1 = src[1];
        }
      } else {
        if (grow < M) {
          const float4* src = (const float4*)((const float*)Aptr + (size_t)grow * K + k0 + kh);
          float4 a = src[0], b = src[1], c = src[2], d = src[3];
          d0.x = packbf2(a.x, a.y); d0.y = packbf2(a.z, a.w);
          d0.z = packbf2(b.x, b.y); d0.w = packbf2(b.z, b.w);
          d1.x = packbf2(c.x, c.y); d1.y = packbf2(c.z, c.w);
          d1.z = packbf2(d.x, d.y); d1.w = packbf2(d.z, d.w);
        }
      }
      *(uint4*)(&As[r * 40 + kh])     = d0;
      *(uint4*)(&As[r * 40 + kh + 8]) = d1;
    }
    // ---- stage B tile 128x32 from Bt (N-major bf16, contiguous in K) ----
    {
      const int r  = tid >> 1;
      const int kh = (tid & 1) << 4;
      const uint4* src = (const uint4*)(Bt + (size_t)(bn + r) * K + k0 + kh);
      uint4 b0 = src[0], b1 = src[1];
      *(uint4*)(&Bs[r * 40 + kh])     = b0;
      *(uint4*)(&Bs[r * 40 + kh + 8]) = b1;
    }
    __syncthreads();

    // ---- fragments per documented CDNA5 layouts ----
    FragAB afr[2], bfr[4];
#pragma unroll
    for (int mt = 0; mt < 2; ++mt) {
      // A 16x32: lane<16 -> row=lm, K 0..7 & 16..23 ; lane>=16 -> K 8..15 & 24..31
      const unsigned short* p = &As[(waveM * 32 + mt * 16 + lm) * 40 + lh * 8];
      afr[mt].q[0] = *(const uint4*)p;
      afr[mt].q[1] = *(const uint4*)(p + 16);
    }
#pragma unroll
    for (int nt = 0; nt < 4; ++nt) {
      // B 32x16: lane<16 -> col=lm, K 0..7 & 16..23 ; lane>=16 -> col=lm, K 8..15 & 24..31
      const unsigned short* p = &Bs[(waveN * 64 + nt * 16 + lm) * 40 + lh * 8];
      bfr[nt].q[0] = *(const uint4*)p;
      bfr[nt].q[1] = *(const uint4*)(p + 16);
    }
#pragma unroll
    for (int mt = 0; mt < 2; ++mt)
#pragma unroll
      for (int nt = 0; nt < 4; ++nt)
        acc[mt][nt] = __builtin_amdgcn_wmma_f32_16x16x32_bf16(
            false, afr[mt].v, false, bfr[nt].v, (short)0, acc[mt][nt], false, false);
    __syncthreads();
  }

  // ---- epilogue: bias + optional tanh; C layout: VGPR j -> row j (+8 for hi lanes), col = lane&15
#pragma unroll
  for (int nt = 0; nt < 4; ++nt) {
    const int col = bn + waveN * 64 + nt * 16 + lm;
    const float bv = bias[col];
#pragma unroll
    for (int mt = 0; mt < 2; ++mt) {
#pragma unroll
      for (int j = 0; j < 8; ++j) {
        const int row = bm + waveM * 32 + mt * 16 + lh * 8 + j;
        if (row < M) {
          float v = acc[mt][nt][j] + bv;
          if (ACT_TANH) v = tanhf(v);
          C[(size_t)row * ldc + col] = v;
        }
      }
    }
  }
}

// ---------------- residual + LayerNorm (+tanh), W=512, one row per block ----------------
template<int DO_TANH>
__global__ __launch_bounds__(256)
void ln_kernel(const float* __restrict__ x, float* __restrict__ h,
               const float* __restrict__ g, const float* __restrict__ be) {
  __shared__ float red[256];
  const size_t row = blockIdx.x;
  const float* xr = x + row * 512;
  float* hr = h + row * 512;
  const int t = threadIdx.x;
  float y0 = xr[t] + hr[t];
  float y1 = xr[t + 256] + hr[t + 256];

  red[t] = y0 + y1;
  __syncthreads();
  for (int o = 128; o > 0; o >>= 1) { if (t < o) red[t] += red[t + o]; __syncthreads(); }
  const float mu = red[0] * (1.0f / 512.0f);
  __syncthreads();
  const float d0 = y0 - mu, d1 = y1 - mu;
  red[t] = d0 * d0 + d1 * d1;
  __syncthreads();
  for (int o = 128; o > 0; o >>= 1) { if (t < o) red[t] += red[t + o]; __syncthreads(); }
  const float rs = rsqrtf(red[0] * (1.0f / 512.0f) + 1e-5f);
  float o0 = d0 * rs * g[t]       + be[t];
  float o1 = d1 * rs * g[t + 256] + be[t + 256];
  if (DO_TANH) { o0 = tanhf(o0); o1 = tanhf(o1); }
  hr[t] = o0;
  hr[t + 256] = o1;
}

// ---------------- host-side encoder weight handles ----------------
struct EncW {
  const unsigned short *W1t, *W2t, *Wpt, *Woutt;
  const float *b1, *b2, *g, *be, *bp, *bout;
  const unsigned short *sW1t[2], *sW2t[2];
  const float *sb1[2], *sb2[2], *sg[2], *sbe[2];
};

extern "C" void kernel_launch(void* const* d_in, const int* in_sizes, int n_in,
                              void* d_out, int out_size, void* d_ws, size_t ws_size,
                              hipStream_t stream) {
  if (n_in < 70) return;
  // Input layout: top-level dict insertion order; "params" flattened in JAX pytree
  // order (dict keys sorted): encoders move(4..23), pick(24..45), transport(46..65).
  const float* agv      = (const float*)d_in[0];
  const float* opE      = (const float*)d_in[1];
  const float* mach     = (const float*)d_in[2];
  const float* wait_emb = (const float*)d_in[3];
  const int* pick_idx   = (const int*)d_in[66];
  const int* trans_idx  = (const int*)d_in[67];
  const int* move_idx   = (const int*)d_in[68];
  float* out = (float*)d_out;

  // ---- workspace layout ----
  char* ws = (char*)d_ws;
  size_t off = 0;
  unsigned short* Xbf = (unsigned short*)ws;       // pick bf16 X: 100000 x 1024
  float* Xf = (float*)ws;                          // trans/move f32 X: 100000 x 512 (same bytes)
  off += (size_t)NPICK * 1024 * 2;
  float* T = (float*)(ws + off); off += (size_t)NPICK * 512 * 4;
  float* H = (float*)(ws + off); off += (size_t)NPICK * 512 * 4;
  float* P = (float*)(ws + off); off += (size_t)NPICK * 512 * 4;
  unsigned short* wtbase = (unsigned short*)(ws + off);
  const size_t WT_ELEMS = 5898240;                 // total bf16 weight elements
  if (ws_size < off + WT_ELEMS * 2) return;

  size_t wcur = 0;
  auto F = [&](int i) -> const float* { return (const float*)d_in[i]; };
  auto trans = [&](int i, int K, int N) -> const unsigned short* {
    unsigned short* dst = wtbase + wcur;
    wcur += (size_t)K * N;
    int total = K * N;
    wtrans_kernel<<<(total + 255) / 256, 256, 0, stream>>>(F(i), dst, K, N);
    return dst;
  };

  // ---- convert + transpose all weights to bf16 (few MB; stays in L2) ----
  EncW mv{}, pk{}, tr{};
  {  // move @ base 4 (no Wp)
    int b = 4;
    mv.Woutt = trans(b + 0, 512, 256); mv.bout = F(b + 1);
    mv.W1t = trans(b + 2, 512, 512);   mv.W2t = trans(b + 3, 512, 512);
    mv.b1 = F(b + 4); mv.b2 = F(b + 5); mv.be = F(b + 6); mv.g = F(b + 7);
    for (int s = 0; s < 2; ++s) {
      int sb = b + 8 + s * 6;
      mv.sW1t[s] = trans(sb + 0, 512, 512); mv.sW2t[s] = trans(sb + 1, 512, 512);
      mv.sb1[s] = F(sb + 2); mv.sb2[s] = F(sb + 3); mv.sbe[s] = F(sb + 4); mv.sg[s] = F(sb + 5);
    }
  }
  {  // pick @ base 24 (has Wp: leaves W1,W2,Wp,b1,b2,be,bp,g)
    int b = 24;
    pk.Woutt = trans(b + 0, 512, 256); pk.bout = F(b + 1);
    pk.W1t = trans(b + 2, 1024, 512);
    pk.W2t = trans(b + 3, 512, 512);
    pk.Wpt = trans(b + 4, 1024, 512);
    pk.b1 = F(b + 5); pk.b2 = F(b + 6); pk.be = F(b + 7); pk.bp = F(b + 8); pk.g = F(b + 9);
    for (int s = 0; s < 2; ++s) {
      int sb = b + 10 + s * 6;
      pk.sW1t[s] = trans(sb + 0, 512, 512); pk.sW2t[s] = trans(sb + 1, 512, 512);
      pk.sb1[s] = F(sb + 2); pk.sb2[s] = F(sb + 3); pk.sbe[s] = F(sb + 4); pk.sg[s] = F(sb + 5);
    }
  }
  {  // transport @ base 46 (no Wp)
    int b = 46;
    tr.Woutt = trans(b + 0, 512, 256); tr.bout = F(b + 1);
    tr.W1t = trans(b + 2, 512, 512);   tr.W2t = trans(b + 3, 512, 512);
    tr.b1 = F(b + 4); tr.b2 = F(b + 5); tr.be = F(b + 6); tr.g = F(b + 7);
    for (int s = 0; s < 2; ++s) {
      int sb = b + 8 + s * 6;
      tr.sW1t[s] = trans(sb + 0, 512, 512); tr.sW2t[s] = trans(sb + 1, 512, 512);
      tr.sb1[s] = F(sb + 2); tr.sb2[s] = F(sb + 3); tr.sbe[s] = F(sb + 4); tr.sg[s] = F(sb + 5);
    }
  }

  auto gemm = [&](const void* A, int a_bf16, const unsigned short* Bt, const float* bias,
                  float* Cp, int M, int N, int K, int ldc, int act) {
    dim3 grid(N / 128, (M + 127) / 128);
    if (a_bf16) {
      if (act) wmma_gemm_kernel<1, 1><<<grid, 256, 0, stream>>>(A, Bt, bias, Cp, M, N, K, ldc);
      else     wmma_gemm_kernel<1, 0><<<grid, 256, 0, stream>>>(A, Bt, bias, Cp, M, N, K, ldc);
    } else {
      if (act) wmma_gemm_kernel<0, 1><<<grid, 256, 0, stream>>>(A, Bt, bias, Cp, M, N, K, ldc);
      else     wmma_gemm_kernel<0, 0><<<grid, 256, 0, stream>>>(A, Bt, bias, Cp, M, N, K, ldc);
    }
  };
  auto ln = [&](const float* x, float* h, const float* g, const float* be, int M) {
    ln_kernel<1><<<M, 256, 0, stream>>>(x, h, g, be);
  };

  auto run_encoder = [&](const EncW& w, const void* X, int a_bf16, int K_in,
                         const float* projf, float* outRows) {
    const int M = NPICK;  // all encoders have 100000 rows
    // in_rl: h = tanh(X@W1+b1)@W2+b2 ; proj = X@Wp+bp or X ; H = tanh(LN(proj+h))
    gemm(X, a_bf16, w.W1t, w.b1, T, M, 512, K_in, 512, 1);
    gemm(T, 0, w.W2t, w.b2, H, M, 512, 512, 512, 0);
    const float* projsrc = projf;
    if (w.Wpt) { gemm(X, a_bf16, w.Wpt, w.bp, P, M, 512, K_in, 512, 0); projsrc = P; }
    ln(projsrc, H, w.g, w.be, M);
    // stack0: X=H -> result in P
    gemm(H, 0, w.sW1t[0], w.sb1[0], T, M, 512, 512, 512, 1);
    gemm(T, 0, w.sW2t[0], w.sb2[0], P, M, 512, 512, 512, 0);
    ln(H, P, w.sg[0], w.sbe[0], M);
    // stack1: X=P -> result in H
    gemm(P, 0, w.sW1t[1], w.sb1[1], T, M, 512, 512, 512, 1);
    gemm(T, 0, w.sW2t[1], w.sb2[1], H, M, 512, 512, 512, 0);
    ln(P, H, w.sg[1], w.sbe[1], M);
    // output projection straight into d_out
    gemm(H, 0, w.Woutt, w.bout, outRows, M, 256, 512, 256, 0);
  };

  // ---- wait section ----
  wait_bcast_kernel<<<NWAIT, 256, 0, stream>>>(wait_emb, out);

  // ---- pick ----
  pick_gather_kernel<<<NPICK, 256, 0, stream>>>(agv, opE, mach, pick_idx, Xbf);
  run_encoder(pk, Xbf, 1, 1024, nullptr, out + (size_t)NWAIT * OUTC);

  // ---- transport ----
  pair_gather_kernel<<<NTRANS / 2, 256, 0, stream>>>(agv, mach, trans_idx, Xf, NTRANS);
  run_encoder(tr, Xf, 0, 512, Xf, out + (size_t)(NWAIT + NPICK) * OUTC);

  // ---- move ----
  pair_gather_kernel<<<NMOVE / 2, 256, 0, stream>>>(agv, mach, move_idx, Xf, NMOVE);
  run_encoder(mv, Xf, 0, 512, Xf, out + (size_t)(NWAIT + NPICK + NTRANS) * OUTC);

  (void)in_sizes; (void)out_size;
}
